// SparseDiscriminator_11063835754776
// MI455X (gfx1250) — matrix operand order repaired and checked by reading, not compiled, not run on hardware
//
#include <hip/hip_runtime.h>
#include <hip/hip_bf16.h>

// ---------------------------------------------------------------------------
// SparseDiscriminator on MI455X (gfx1250, wave32, WMMA + TDM)
//
// sparse conv = TDM gather (tensor_load_to_lds, gather_mode) -> LDS ->
// v_wmma_f32_16x16x32_bf16 -> f32 atomic scatter.
// Working set is L2-resident (<=26MB/level vs 192MB L2); compute is tiny,
// so the design targets the gather path: hardware DMA gather of the 16
// indexed rows per tile + DMA of the cin x 16 W[k] tile, TENSORcnt-tracked.
// ---------------------------------------------------------------------------

typedef __attribute__((ext_vector_type(16))) __bf16 v16bf;
typedef __attribute__((ext_vector_type(8)))  float  v8f;
typedef unsigned int v4u __attribute__((ext_vector_type(4)));
typedef int          v8i __attribute__((ext_vector_type(8)));
typedef int          v4i __attribute__((ext_vector_type(4)));

#define K3 27

__device__ __forceinline__ unsigned short f32_to_bf16(float f) {
  union { float f; unsigned u; } v; v.f = f;
  unsigned r = v.u + 0x7FFFu + ((v.u >> 16) & 1u);   // round-to-nearest-even
  return (unsigned short)(r >> 16);
}

__device__ __forceinline__ unsigned f32_key(float f) {
  // order-preserving monotone map float -> uint (for atomicMax-based segmax)
  union { float f; unsigned u; } v; v.f = f;
  return (v.u & 0x80000000u) ? ~v.u : (v.u | 0x80000000u);
}

// Issue one TDM descriptor (D# groups 0..3). ROCm7.2 clang-22 = 5-arg builtin,
// amdgpu-toolchain clang-23 = 6-arg builtin.
__device__ __forceinline__ void tdm_load(v4u g0, v8i g1, v4i g2, v4i g3) {
#if __clang_major__ >= 23
  __builtin_amdgcn_tensor_load_to_lds(g0, g1, g2, g3, (v8i){0,0,0,0,0,0,0,0}, 0);
#else
  __builtin_amdgcn_tensor_load_to_lds(g0, g1, g2, g3, 0);
#endif
}

__global__ void zero_ws_kernel(float* __restrict__ p, long n) {
  long i = (long)blockIdx.x * blockDim.x + threadIdx.x;
  long stride = (long)gridDim.x * blockDim.x;
  for (; i < n; i += stride) p[i] = 0.0f;
}

__global__ void init_consts_kernel(int* __restrict__ p, int v) {
  if (threadIdx.x == 0) p[0] = v;
}

// ---------------------------------------------------------------------------
// Sparse conv: one wave per (offset k, 16-pair tile, 16-cout tile).
// grid.x = 27 * ceil(P/16), grid.y = cout/16, block = 32 (EXEC all-1s at WMMA).
// A (16 gathered rows x cin) arrives via two TDM gather descriptors
// (32-bit indices, 8 rows each; tensor_dim1 = n_in so pad rows read zero).
// B (cin x 16 slice of W[k]) arrives via one plain 2D TDM descriptor.
// ---------------------------------------------------------------------------
__global__ __launch_bounds__(32) void sparse_conv_wmma_kernel(
    const float* __restrict__ x,        // n_in rows x cin
    const float* __restrict__ W,        // K3 x cin x cout
    const int*   __restrict__ in_idx,   // K3 x P  (pad value == n_in)
    const int*   __restrict__ out_idx,  // K3 x P  (pad value == n_out)
    const int*   __restrict__ n_in_p,   // device scalar: active input rows
    const int*   __restrict__ n_out_p,  // device scalar: active output rows
    float*       __restrict__ out,      // (bound+1) x cout, pre-zeroed
    int cin, int cout, int P, int PT)
{
  __shared__ float smemA[16 * 256];     // 16 gathered rows x cin (row stride = cin)
  __shared__ float smemB[256 * 16];     // cin x 16 weight tile  (row stride = 16)
  __shared__ int   s_out[16];

  const int k    = blockIdx.x / PT;
  const int tile = blockIdx.x - k * PT;
  const int pairBase = tile * 16;
  const int lane = threadIdx.x;
  const int hl   = lane >> 4;           // lane half (ISA 7.12.2 layouts)
  const int mn   = lane & 15;

  const int n_in  = *n_in_p;
  const int n_out = *n_out_p;

  // out indices: per-lane load -> LDS (needed lane-indexed in the epilogue)
  if (lane < 16) {
    int p = pairBase + lane;
    s_out[lane] = (p < P) ? out_idx[(long)k * P + p] : 0x7FFFFFFF;
  }

  // in indices: uniform scalar loads (feed SGPR-resident TDM descriptors)
  const long ibase = (long)k * P + pairBase;
  int ridx[16];
  #pragma unroll
  for (int i = 0; i < 16; ++i)
    ridx[i] = (pairBase + i < P) ? in_idx[ibase + i] : n_in;   // pad -> OOB row

  const unsigned ldsA = (unsigned)(unsigned long long)(void*)smemA;
  const unsigned ldsB = (unsigned)(unsigned long long)(void*)smemB;
  const int ng0 = blockIdx.y * 16;
  const float* Wk = W + (long)k * cin * cout;

  // ---- TDM gather of A rows: D# group0/1 + 8 row indices in groups 2/3 ----
  {
    const unsigned long long xaddr = (unsigned long long)(const void*)x;
    v4u g0 = { 1u /*count*/ | (1u << 30) /*32b idx*/ | (1u << 31) /*gather*/,
               ldsA,
               (unsigned)xaddr,
               (unsigned)((xaddr >> 32) & 0x01FFFFFFull) | (2u << 30) /*type=image*/ };
    v8i g1 = { (int)0x20000u,                                        // data_size=4B
               (int)(((unsigned)cin & 0xFFFFu) << 16),               // tensor_dim0.lo
               (int)((((unsigned)cin >> 16) & 0xFFFFu) |
                     (((unsigned)n_in & 0xFFFFu) << 16)),            // dim0.hi | dim1.lo (=n_in -> OOB rows read 0)
               (int)((((unsigned)n_in >> 16) & 0xFFFFu) |
                     (((unsigned)cin & 0xFFFFu) << 16)),             // dim1.hi | tile_dim0 (=cin)
               8,                                                    // tile_dim1 = #valid indices
               (int)cin,                                             // tensor_dim0_stride.lo
               0, 0 };
    v4i i0 = { ridx[0], ridx[1], ridx[2],  ridx[3]  };
    v4i i1 = { ridx[4], ridx[5], ridx[6],  ridx[7]  };
    v4i i2 = { ridx[8], ridx[9], ridx[10], ridx[11] };
    v4i i3 = { ridx[12], ridx[13], ridx[14], ridx[15] };
    tdm_load(g0, g1, i0, i1);                       // rows 0..7  -> smemA[0 ..)
    v4u g0b = g0;
    g0b.y = ldsA + 8u * (unsigned)cin * 4u;         // rows 8..15 -> smemA[8*cin ..)
    tdm_load(g0b, g1, i2, i3);
  }

  // ---- TDM 2D tile load of B = W[k][0:cin, ng0:ng0+16] --------------------
  {
    const unsigned long long baddr = (unsigned long long)(const void*)(Wk + ng0);
    v4u g0 = { 1u, ldsB,
               (unsigned)baddr,
               (unsigned)((baddr >> 32) & 0x01FFFFFFull) | (2u << 30) };
    v8i g1 = { (int)0x20000u,                                        // data_size=4B
               (int)(16u << 16),                                     // tensor_dim0 = 16
               (int)(((unsigned)cin & 0xFFFFu) << 16),               // dim0.hi(0) | dim1.lo = cin
               (int)((((unsigned)cin >> 16) & 0xFFFFu) | (16u << 16)), // dim1.hi | tile_dim0 = 16
               (int)cin,                                             // tile_dim1 = cin
               (int)cout,                                            // tensor_dim0_stride = cout
               0, 0 };
    tdm_load(g0, g1, (v4i){0,0,0,0}, (v4i){0,0,0,0});
  }

  __builtin_amdgcn_s_wait_tensorcnt(0);   // TDM -> LDS complete
  __syncthreads();

  const int ng = ng0 + mn;                // B/C/D: N = lane%16 (output channel)
  v8f acc = {};
  const int ksteps = (cin + 31) >> 5;

  for (int s = 0; s < ksteps; ++s) {
    // A fragment: 16x32 bf16. Lane l (half h, m=l%16): VGPR j holds
    // K = (j<4?0:16) + 8*h + 2*(j&3) and +1  (ISA p.108 16-bit A 16x32).
    union { unsigned short h[16]; v16bf v; } A;
    #pragma unroll
    for (int j = 0; j < 8; ++j) {
      int kk = ((j < 4) ? 0 : 16) + hl * 8 + (j & 3) * 2;
      int c0 = (s << 5) + kk;
      float f0 = (c0     < cin) ? smemA[mn * cin + c0]     : 0.0f;
      float f1 = (c0 + 1 < cin) ? smemA[mn * cin + c0 + 1] : 0.0f;
      A.h[2 * j]     = f32_to_bf16(f0);
      A.h[2 * j + 1] = f32_to_bf16(f1);
    }
    // B fragment: 32x16 bf16. Lane l (half h, n=l%16): VGPR j holds
    // K = (j<4?0:16) + 8*h + 2*(j&3) and +1 (same K order as A).
    union { unsigned short h[16]; v16bf v; } Bm;
    #pragma unroll
    for (int j = 0; j < 8; ++j) {
      int c0 = (s << 5) + ((j < 4) ? 0 : 16) + hl * 8 + (j & 3) * 2;
      float f0 = (c0     < cin) ? smemB[c0 * 16 + mn]       : 0.0f;
      float f1 = (c0 + 1 < cin) ? smemB[(c0 + 1) * 16 + mn] : 0.0f;
      Bm.h[2 * j]     = f32_to_bf16(f0);
      Bm.h[2 * j + 1] = f32_to_bf16(f1);
    }
    acc = __builtin_amdgcn_wmma_f32_16x16x32_bf16(
        /*neg_a=*/false, A.v, /*neg_b=*/false, Bm.v,
        /*c_mod=*/(short)0, acc, /*reuse_a=*/false, /*reuse_b=*/false);
  }

  // C/D layout: lanes 0-15 VGPR r -> M=r, lanes 16-31 -> M=r+8 (ISA p.108).
  #pragma unroll
  for (int r = 0; r < 8; ++r) {
    int m    = r + 8 * hl;
    int orow = s_out[m];
    if (orow < n_out)     // drops the reference's trash row + padded pairs
      atomicAdd(&out[(long)orow * cout + ng], acc[r]);
  }
}

// ---------------------------------------------------------------------------
// BatchNorm pass 1: per-channel sum & sumsq (wave shuffle + LDS + atomics).
// ---------------------------------------------------------------------------
__global__ __launch_bounds__(256) void bn_stats_kernel(
    const float* __restrict__ x, const int* __restrict__ n_p,
    int cin, float* __restrict__ stats /* [2*cin], pre-zeroed */)
{
  __shared__ float ls[8], ls2[8];
  const int  n   = *n_p;
  const int  tid = threadIdx.x;
  const long row = (long)blockIdx.x * 256 + tid;
  const bool on  = (row < n);
  for (int c = 0; c < cin; ++c) {
    float v = on ? x[row * cin + c] : 0.0f;
    float s = v, s2 = v * v;
    #pragma unroll
    for (int off = 16; off >= 1; off >>= 1) {
      s  += __shfl_down(s,  off, 32);
      s2 += __shfl_down(s2, off, 32);
    }
    if ((tid & 31) == 0) { ls[tid >> 5] = s; ls2[tid >> 5] = s2; }
    __syncthreads();
    if (tid == 0) {
      float t = 0.0f, t2 = 0.0f;
      #pragma unroll
      for (int i = 0; i < 8; ++i) { t += ls[i]; t2 += ls2[i]; }
      atomicAdd(&stats[c], t);
      atomicAdd(&stats[cin + c], t2);
    }
    __syncthreads();
  }
}

// BatchNorm pass 2 + LeakyReLU(0.2), in place.
__global__ void bn_apply_lrelu_kernel(
    float* __restrict__ x, const float* __restrict__ stats,
    const float* __restrict__ g, const float* __restrict__ b,
    const int* __restrict__ n_p, int cin)
{
  const int n = *n_p;
  long idx = (long)blockIdx.x * blockDim.x + threadIdx.x;
  if (idx >= (long)n * cin) return;
  int   c     = (int)(idx % cin);
  float inv_n = 1.0f / (float)n;
  float mu    = stats[c] * inv_n;
  float var   = stats[cin + c] * inv_n - mu * mu;   // biased var (matches ref)
  float y = (x[idx] - mu) * rsqrtf(var + 1e-5f) * g[c] + b[c];
  x[idx] = (y >= 0.0f) ? y : 0.2f * y;
}

// Per-batch global max pool via order-preserving uint atomicMax.
__global__ void segmax_kernel(
    const float* __restrict__ x, const int* __restrict__ bidx,
    unsigned* __restrict__ pooled /* 4*cin, pre-zeroed */, int nrows, int cin)
{
  long idx = (long)blockIdx.x * blockDim.x + threadIdx.x;
  if (idx >= (long)nrows * cin) return;
  int row = (int)(idx / cin), c = (int)(idx % cin);
  atomicMax(&pooled[bidx[row] * cin + c], f32_key(x[idx]));
}

// Final 256 -> 1 linear per batch.
__global__ __launch_bounds__(256) void final_linear_kernel(
    const unsigned* __restrict__ pooled, const float* __restrict__ Wf,
    const float* __restrict__ bf, float* __restrict__ out, int cin)
{
  __shared__ float ls[8];
  const int b = blockIdx.x, tid = threadIdx.x;
  unsigned u    = pooled[b * cin + tid];
  unsigned bits = (u & 0x80000000u) ? (u & 0x7FFFFFFFu) : ~u;  // inverse key
  union { unsigned u; float f; } v; v.u = bits;
  float s = v.f * Wf[tid];
  #pragma unroll
  for (int off = 16; off >= 1; off >>= 1) s += __shfl_down(s, off, 32);
  if ((tid & 31) == 0) ls[tid >> 5] = s;
  __syncthreads();
  if (tid == 0) {
    float t = 0.0f;
    #pragma unroll
    for (int i = 0; i < 8; ++i) t += ls[i];
    out[b] = t + bf[0];
  }
}

// ---------------------------------------------------------------------------
// Host orchestration. d_in flattening assumed in setup_inputs() insertion
// order: feats | 5x{W1,g1,b1,W2,g2,b2} | Wf | bf | 5x{in1,out1,n1,in2,out2,n2}
// | batch_index  (n_in == 64). n_out per level is data-dependent -> read on
// device from the scalar inputs; host grids use analytic upper bounds.
// ---------------------------------------------------------------------------
extern "C" void kernel_launch(void* const* d_in, const int* in_sizes, int n_in,
                              void* d_out, int out_size, void* d_ws, size_t ws_size,
                              hipStream_t stream)
{
  (void)out_size; (void)ws_size;
  if (n_in < 64) return;

  static const int  CH[5]    = {16, 32, 64, 128, 256};
  static const long BOUND[5] = {400000, 131072, 16384, 2048, 256}; // 4*(128/2^(l+1))^3 capped

  const float* feats = (const float*)d_in[0];
  const float *W1[5], *g1[5], *b1[5], *W2[5], *g2[5], *b2[5];
  for (int l = 0; l < 5; ++l) {
    W1[l] = (const float*)d_in[1 + 6 * l + 0];
    g1[l] = (const float*)d_in[1 + 6 * l + 1];
    b1[l] = (const float*)d_in[1 + 6 * l + 2];
    W2[l] = (const float*)d_in[1 + 6 * l + 3];
    g2[l] = (const float*)d_in[1 + 6 * l + 4];
    b2[l] = (const float*)d_in[1 + 6 * l + 5];
  }
  const float* Wf  = (const float*)d_in[31];
  const float* bfp = (const float*)d_in[32];
  const int *in1[5], *out1[5], *n1[5], *in2[5], *out2[5], *n2[5];
  int P1[5], P2[5];
  for (int l = 0; l < 5; ++l) {
    in1[l]  = (const int*)d_in[33 + 6 * l + 0];
    out1[l] = (const int*)d_in[33 + 6 * l + 1];
    n1[l]   = (const int*)d_in[33 + 6 * l + 2];
    in2[l]  = (const int*)d_in[33 + 6 * l + 3];
    out2[l] = (const int*)d_in[33 + 6 * l + 4];
    n2[l]   = (const int*)d_in[33 + 6 * l + 5];
    P1[l]   = in_sizes[33 + 6 * l + 0] / K3;
    P2[l]   = in_sizes[33 + 6 * l + 3] / K3;
  }
  const int* bidx = (const int*)d_in[63];
  const int  n5   = in_sizes[63];

  // ---- workspace carve (all zeroed each call: determinism + atomic accum) --
  float* wsf = (float*)d_ws;
  long off = 0;
  float *y1[5], *y2[5];
  for (int l = 0; l < 5; ++l) {
    y1[l] = wsf + off; off += (BOUND[l] + 1) * CH[l];
    y2[l] = wsf + off; off += (BOUND[l] + 1) * CH[l];
  }
  float*    stats  = wsf + off; off += 10 * 512;     // 10 BNs x (sum|sumsq) x 256
  unsigned* pooled = (unsigned*)(wsf + off); off += 1024;
  int*      cN0    = (int*)(wsf + off); off += 16;
  const long totalF = off;                            // ~96 MB of f32 scratch

  zero_ws_kernel<<<2048, 256, 0, stream>>>(wsf, totalF);
  init_consts_kernel<<<1, 32, 0, stream>>>(cN0, 400000);

  const float* xin  = feats;
  const int*   ninp = cN0;
  int cin = 1;
  for (int l = 0; l < 5; ++l) {
    const int cout = CH[l];
    // stride-2 conv
    {
      int PT = (P1[l] + 15) / 16;
      dim3 grid(K3 * PT, cout / 16);
      sparse_conv_wmma_kernel<<<grid, 32, 0, stream>>>(
          xin, W1[l], in1[l], out1[l], ninp, n1[l], y1[l], cin, cout, P1[l], PT);
    }
    float* st = stats + (2 * l) * 512;
    bn_stats_kernel<<<(unsigned)((BOUND[l] + 255) / 256), 256, 0, stream>>>(y1[l], n1[l], cout, st);
    long tot = BOUND[l] * (long)cout;
    bn_apply_lrelu_kernel<<<(unsigned)((tot + 255) / 256), 256, 0, stream>>>(
        y1[l], st, g1[l], b1[l], n1[l], cout);
    // stride-1 conv at new tensor stride
    {
      int PT = (P2[l] + 15) / 16;
      dim3 grid(K3 * PT, cout / 16);
      sparse_conv_wmma_kernel<<<grid, 32, 0, stream>>>(
          y1[l], W2[l], in2[l], out2[l], n1[l], n2[l], y2[l], cout, cout, P2[l], PT);
    }
    float* st2 = stats + (2 * l + 1) * 512;
    bn_stats_kernel<<<(unsigned)((BOUND[l] + 255) / 256), 256, 0, stream>>>(y2[l], n2[l], cout, st2);
    bn_apply_lrelu_kernel<<<(unsigned)((tot + 255) / 256), 256, 0, stream>>>(
        y2[l], st2, g2[l], b2[l], n2[l], cout);
    xin = y2[l]; ninp = n2[l]; cin = cout;
  }

  long segtot = (long)n5 * 256;
  segmax_kernel<<<(unsigned)((segtot + 255) / 256), 256, 0, stream>>>(y2[4], bidx, pooled, n5, 256);
  final_linear_kernel<<<4, 256, 0, stream>>>(pooled, Wf, bfp, (float*)d_out, 256);
}
